// MultiHeadAttention_61237643707046
// MI455X (gfx1250) — matrix operand order repaired and checked by reading, not compiled, not run on hardware
//
#include <hip/hip_runtime.h>
#include <hip/hip_bf16.h>

// ---------------------------------------------------------------------------
// MHA block for MI455X (gfx1250): bf16 WMMA everywhere, single-touch attn
// output, flash-style online softmax stats with score recomputation.
// WMMA fragments are ds_load_b128 pairs; contiguous tile copies use the
// CDNA5 async global->LDS path (ASYNCcnt) when the toolchain exposes it.
// ---------------------------------------------------------------------------

typedef __attribute__((ext_vector_type(16))) __bf16        v16bf;
typedef __attribute__((ext_vector_type(8)))  float         v8f;
typedef __attribute__((ext_vector_type(4)))  unsigned int  v4u;

#define BATCH  4
#define S_LEN  2048
#define DMODEL 1024
#define NH     16
#define HD     64
#define LN_EPS 1e-5f

// ---------------- CDNA5 async global->LDS (ASYNCcnt) path ------------------
#if defined(__AMDGCN__) && __has_builtin(__builtin_amdgcn_global_load_async_to_lds_b128)
#define HAVE_ASYNC_LDS 1
// Builtin signature (from hipcc diagnostic): param0 is a pointer to a GCC
// vector "int __attribute__((vector_size(16)))" in AS1 (__device__), param1
// the LDS-side pointer in AS3 (__shared__), then imm offset and imm cpol.
typedef int v4i_g __attribute__((vector_size(16)));
typedef __attribute__((address_space(1))) v4i_g* as1_v4i;
typedef __attribute__((address_space(3))) v4i_g* as3_v4i;
__device__ __forceinline__ void async_ld_b128(const void* g, void* l) {
  __builtin_amdgcn_global_load_async_to_lds_b128((as1_v4i)g, (as3_v4i)l, 0, 0);
}
__device__ __forceinline__ void async_wait0() {
#if __has_builtin(__builtin_amdgcn_s_wait_asynccnt)
  __builtin_amdgcn_s_wait_asynccnt(0);
#else
  asm volatile("s_wait_asynccnt 0x0" ::: "memory");
#endif
}
#else
#define HAVE_ASYNC_LDS 0
#endif

__device__ __forceinline__ __bf16 f2bf(float f) { return (__bf16)f; }

union Frag32B { v4u u[2]; v16bf bf; };
union BfPair  { __bf16 h[2]; unsigned int u; };

__device__ __forceinline__ v8f zero_v8f() {
  v8f z;
#pragma unroll
  for (int i = 0; i < 8; ++i) z[i] = 0.0f;
  return z;
}

__device__ __forceinline__ v8f wmma_bf16(v16bf a, v16bf b, v8f c) {
  // D = A(16x32 bf16) * B(32x16 bf16) + C(16x16 f32)
  return __builtin_amdgcn_wmma_f32_16x16x32_bf16(
      /*neg_a=*/false, a, /*neg_b=*/false, b,
      /*c_mod=*/(short)0, c, /*reuse_a=*/false, /*reuse_b=*/false);
}

// Two 16-byte vector loads -> one 32-byte fragment.
__device__ __forceinline__ v16bf ld_frag2(const __bf16* p0, const __bf16* p1) {
  Frag32B f;
  f.u[0] = *(const v4u*)p0;
  f.u[1] = *(const v4u*)p1;
  return f.bf;
}

// A-fragment (16x32 bf16, MxK) from row-major LDS tile [rows][ld].
// ISA layout: lane L (M = L%16); halfword e: K = 16*(e/8) + 8*(L/16) + (e%8)
// => two contiguous 8-element runs at p and p+16.  Requires ld%8==0, kBase%16==0.
__device__ __forceinline__ v16bf frag_a(const __bf16* lds, int ld, int mBase,
                                        int kBase, int lane) {
  const __bf16* p = lds + (mBase + (lane & 15)) * ld + kBase + ((lane >> 4) << 3);
  return ld_frag2(p, p + 16);
}

// B-fragment (32x16 bf16, KxN) from N-major (transposed) LDS tile [N][ld]:
// element B[k][n] = ldsT[n*ld + k].  ISA layout: lane L (N = L%16);
// halfword e: K = 16*(e/8) + 8*(L/16) + (e%8) => two 8-element runs at p and p+16.
__device__ __forceinline__ v16bf frag_b(const __bf16* ldsT, int ld, int nBase,
                                        int kBase, int lane) {
  const __bf16* p = ldsT + (nBase + (lane & 15)) * ld + kBase + ((lane >> 4) << 3);
  return ld_frag2(p, p + 16);
}

// Contiguous global(bf16) -> LDS(bf16) tile copy, nBytes multiple of 4096.
__device__ __forceinline__ void tile_copy(const __bf16* g, __bf16* l, int nBytes,
                                          int tid) {
#if HAVE_ASYNC_LDS
  const char* gs = (const char*)g;
  char* ls = (char*)l;
  const int chunks = nBytes >> 4;
  for (int i = tid; i < chunks; i += 256)
    async_ld_b128(gs + i * 16, ls + i * 16);
#else
  const v4u* src = (const v4u*)g;
  v4u* dst = (v4u*)l;
  const int chunks = nBytes >> 4;
  for (int i = tid; i < chunks; i += 256) dst[i] = src[i];
#endif
}

__device__ __forceinline__ void tile_copy_fence() {
#if HAVE_ASYNC_LDS
  async_wait0();
#endif
}

// ---------------------------------------------------------------------------
// Kernel 1: fused QKV projections.  blockIdx.z selects Q/K/V.
// Output layout: [B*H][S][HD] bf16 (head-major for the attention kernel).
// ---------------------------------------------------------------------------
__global__ __launch_bounds__(256) void qkv_proj_kernel(
    const float* __restrict__ inQ, const float* __restrict__ inK,
    const float* __restrict__ inV, const float* __restrict__ WQ,
    const float* __restrict__ WK, const float* __restrict__ WV,
    __bf16* __restrict__ Qbf, __bf16* __restrict__ Kbf,
    __bf16* __restrict__ Vbf) {
  __shared__ __attribute__((aligned(16))) __bf16 As[128 * 32];  // [m][k]
  __shared__ __attribute__((aligned(16))) __bf16 Bs[128 * 32];  // [n][k] (transposed)
  const int tid = threadIdx.x, lane = tid & 31, wave = tid >> 5;
  const int rowBase = blockIdx.x * 128;
  const int colBase = blockIdx.y * 128;

  const float* A;
  const float* W;
  __bf16* O;
  if (blockIdx.z == 0)      { A = inQ; W = WQ; O = Qbf; }
  else if (blockIdx.z == 1) { A = inK; W = WK; O = Kbf; }
  else                      { A = inV; W = WV; O = Vbf; }

  v8f acc[8];
#pragma unroll
  for (int i = 0; i < 8; ++i) acc[i] = zero_v8f();

  for (int kk0 = 0; kk0 < DMODEL; kk0 += 32) {
    if (kk0 + 32 < DMODEL)
      __builtin_prefetch(&A[(size_t)(rowBase + (tid >> 1)) * DMODEL + kk0 + 32], 0, 1);
    // A tile: each thread converts one contiguous 16-float run -> 2 b128 stores
    {
      const int t16 = tid * 16;                 // 256*16 == 128*32
      const int r = t16 >> 5, c = t16 & 31;
      const float* src = &A[(size_t)(rowBase + r) * DMODEL + kk0 + c];
      __attribute__((aligned(16))) __bf16 tmp[16];
#pragma unroll
      for (int j = 0; j < 16; ++j) tmp[j] = f2bf(src[j]);
      *(v4u*)&As[t16]     = *(const v4u*)&tmp[0];
      *(v4u*)&As[t16 + 8] = *(const v4u*)&tmp[8];
    }
    // B tile, stored transposed [n][k]; pack row pairs into b32 stores
    for (int i = tid; i < 2048; i += 256) {
      const int rp = i >> 7, c = i & 127;       // consecutive threads -> consecutive c
      const int r = 2 * rp;
      BfPair pk;
      pk.h[0] = f2bf(W[(size_t)(kk0 + r)     * DMODEL + colBase + c]);
      pk.h[1] = f2bf(W[(size_t)(kk0 + r + 1) * DMODEL + colBase + c]);
      *(unsigned int*)&Bs[c * 32 + r] = pk.u;
    }
    __syncthreads();
    v16bf a = frag_a(As, 32, wave * 16, 0, lane);
#pragma unroll
    for (int nt = 0; nt < 8; ++nt) {
      v16bf b = frag_b(Bs, 32, nt * 16, 0, lane);
      acc[nt] = wmma_bf16(a, b, acc[nt]);
    }
    __syncthreads();
  }

  const int hi = lane >> 4, nn = lane & 15;
#pragma unroll
  for (int nt = 0; nt < 8; ++nt) {
#pragma unroll
    for (int r = 0; r < 8; ++r) {
      int R = rowBase + wave * 16 + r + 8 * hi;   // global row = b*S + s
      int C = colBase + nt * 16 + nn;             // global col = h*HD + d
      int bb = R >> 11, s = R & (S_LEN - 1);
      int h = C >> 6, d = C & (HD - 1);
      O[(((size_t)(bb * NH + h)) * S_LEN + s) * HD + d] = f2bf(acc[nt][r]);
    }
  }
}

// ---------------------------------------------------------------------------
// Kernel 2: causal attention per (b*H+h, 128-row q-tile).
//  Pass A: online (m,l) stats via WMMA QK^T tiles (scores discarded).
//  Pass B: recompute scores, write attn probabilities ONCE, p@V via WMMA.
// ---------------------------------------------------------------------------
__global__ __launch_bounds__(256) void attention_kernel(
    const __bf16* __restrict__ Qbf, const __bf16* __restrict__ Kbf,
    const __bf16* __restrict__ Vbf, float* __restrict__ attn,
    __bf16* __restrict__ Ctx) {
  __shared__ __attribute__((aligned(16))) __bf16 Qs[128 * HD];    // [q][d]   16 KB
  __shared__ __attribute__((aligned(16))) __bf16 KPs[128 * 128];  // K [key][d] / p [q][key] 32 KB
  __shared__ __attribute__((aligned(16))) __bf16 Vs[HD * 128];    // [dv][key] 16 KB

  const int tid = threadIdx.x, lane = tid & 31, wave = tid >> 5;
  const int bh = blockIdx.x;          // b*NH + h
  const int qt = blockIdx.y;
  const int qBase = qt * 128;
  const int hi = lane >> 4, nn = lane & 15;
  const float scale = 0.125f;         // 1/sqrt(64)

  // Load Q tile once (async contiguous copy).
  tile_copy(Qbf + ((size_t)bh * S_LEN + qBase) * HD, Qs, 128 * HD * 2, tid);
  tile_copy_fence();
  __syncthreads();

  const v16bf a0 = frag_a(Qs, HD, wave * 16, 0, lane);
  const v16bf a1 = frag_a(Qs, HD, wave * 16, 32, lane);

  float m[8], l[8];
#pragma unroll
  for (int r = 0; r < 8; ++r) { m[r] = -3.0e38f; l[r] = 0.0f; }

  // ------------------------- PASS A: softmax stats -------------------------
  for (int kt = 0; kt <= qt; ++kt) {
    tile_copy(Kbf + ((size_t)bh * S_LEN + kt * 128) * HD, KPs, 128 * HD * 2, tid);
    if (kt < qt)
      __builtin_prefetch(Kbf + ((size_t)bh * S_LEN + (kt + 1) * 128) * HD + tid * 32, 0, 1);
    tile_copy_fence();
    __syncthreads();

    v8f sc[8];
#pragma unroll
    for (int nt = 0; nt < 8; ++nt) {
      sc[nt] = zero_v8f();
      v16bf b0 = frag_b(KPs, HD, nt * 16, 0, lane);    // K[key][d] is N-major for QK^T
      sc[nt] = wmma_bf16(a0, b0, sc[nt]);
      v16bf b1 = frag_b(KPs, HD, nt * 16, 32, lane);
      sc[nt] = wmma_bf16(a1, b1, sc[nt]);
    }
    __syncthreads();

#pragma unroll
    for (int r = 0; r < 8; ++r) {
      const int q = qBase + wave * 16 + r + 8 * hi;
      float rowm = -3.0e38f;
#pragma unroll
      for (int nt = 0; nt < 8; ++nt) {
        int k = kt * 128 + nt * 16 + nn;
        float s = sc[nt][r] * scale;
        if (k > q) s = -1.0e9f;
        sc[nt][r] = s;
        rowm = fmaxf(rowm, s);
      }
#pragma unroll
      for (int off = 1; off < 16; off <<= 1)
        rowm = fmaxf(rowm, __shfl_xor(rowm, off, 32));
      float newm = fmaxf(m[r], rowm);
      float rs = 0.0f;
#pragma unroll
      for (int nt = 0; nt < 8; ++nt) rs += __expf(sc[nt][r] - newm);
#pragma unroll
      for (int off = 1; off < 16; off <<= 1) rs += __shfl_xor(rs, off, 32);
      l[r] = l[r] * __expf(m[r] - newm) + rs;
      m[r] = newm;
    }
  }

  float linv[8];
#pragma unroll
  for (int r = 0; r < 8; ++r) linv[r] = 1.0f / l[r];

  v8f ctx[4];
#pragma unroll
  for (int dt = 0; dt < 4; ++dt) ctx[dt] = zero_v8f();

  // -------------------- PASS B: attn write + p @ V -------------------------
  for (int kt = 0; kt <= qt; ++kt) {
    // K tile async; V tile transposed to [dv][key] manually (overlaps async)
    tile_copy(Kbf + ((size_t)bh * S_LEN + kt * 128) * HD, KPs, 128 * HD * 2, tid);
    {
      const __bf16* vbase = Vbf + ((size_t)bh * S_LEN + kt * 128) * HD;
      for (int i = tid; i < 4096; i += 256) {
        const int dv = i & 63, kp = i >> 6;   // consecutive threads -> consecutive dv
        const int key = 2 * kp;
        BfPair pk;
        pk.h[0] = vbase[(size_t)key * HD + dv];
        pk.h[1] = vbase[(size_t)(key + 1) * HD + dv];
        *(unsigned int*)&Vs[dv * 128 + key] = pk.u;
      }
    }
    tile_copy_fence();
    __syncthreads();

    v8f sc[8];
#pragma unroll
    for (int nt = 0; nt < 8; ++nt) {
      sc[nt] = zero_v8f();
      v16bf b0 = frag_b(KPs, HD, nt * 16, 0, lane);
      sc[nt] = wmma_bf16(a0, b0, sc[nt]);
      v16bf b1 = frag_b(KPs, HD, nt * 16, 32, lane);
      sc[nt] = wmma_bf16(a1, b1, sc[nt]);
    }
    __syncthreads();  // done reading KPs as K; reuse as p tile

#pragma unroll
    for (int r = 0; r < 8; ++r) {
      const int q = qBase + wave * 16 + r + 8 * hi;
      const int prow = wave * 16 + r + 8 * hi;
#pragma unroll
      for (int nt = 0; nt < 8; ++nt) {
        int k = kt * 128 + nt * 16 + nn;
        float s = sc[nt][r] * scale;
        float p = (k > q) ? 0.0f : __expf(s - m[r]) * linv[r];
        attn[((size_t)bh * S_LEN + q) * S_LEN + k] = p;
        KPs[prow * 128 + nt * 16 + nn] = f2bf(p);
      }
    }
    __syncthreads();  // p tile complete

    v16bf pa[4];
#pragma unroll
    for (int kkt = 0; kkt < 4; ++kkt)
      pa[kkt] = frag_a(KPs, 128, wave * 16, kkt * 32, lane);
#pragma unroll
    for (int dt = 0; dt < 4; ++dt) {
#pragma unroll
      for (int kkt = 0; kkt < 4; ++kkt) {
        v16bf bv = frag_b(Vs, 128, dt * 16, kkt * 32, lane);  // [dv][key] N-major
        ctx[dt] = wmma_bf16(pa[kkt], bv, ctx[dt]);
      }
    }
    __syncthreads();  // before next iteration overwrites KPs/Vs
  }

  // Zero-fill the masked (strictly upper-triangular) attn columns.
  {
    const int startK = (qt + 1) * 128;
    const int width = S_LEN - startK;
    for (int idx = tid; idx < 128 * width; idx += 256) {
      int rr = idx / width, cc = idx - rr * width;
      attn[((size_t)bh * S_LEN + qBase + rr) * S_LEN + startK + cc] = 0.0f;
    }
  }

  // Store context bf16 in [B, S, H*HD] layout for the output projection.
  const int b = bh / NH, h = bh - b * NH;
#pragma unroll
  for (int dt = 0; dt < 4; ++dt) {
#pragma unroll
    for (int r = 0; r < 8; ++r) {
      int q = qBase + wave * 16 + r + 8 * hi;
      int col = h * HD + dt * 16 + nn;
      Ctx[((size_t)b * S_LEN + q) * DMODEL + col] = f2bf(ctx[dt][r]);
    }
  }
}

// ---------------------------------------------------------------------------
// Kernel 3: output projection + residual -> fp32 x (workspace).
// ---------------------------------------------------------------------------
__global__ __launch_bounds__(256) void out_proj_kernel(
    const __bf16* __restrict__ Actx, const float* __restrict__ WO,
    const float* __restrict__ resid, float* __restrict__ X) {
  __shared__ __attribute__((aligned(16))) __bf16 As[128 * 32];  // [m][k]
  __shared__ __attribute__((aligned(16))) __bf16 Bs[128 * 32];  // [n][k]
  const int tid = threadIdx.x, lane = tid & 31, wave = tid >> 5;
  const int rowBase = blockIdx.x * 128;
  const int colBase = blockIdx.y * 128;

  v8f acc[8];
#pragma unroll
  for (int i = 0; i < 8; ++i) acc[i] = zero_v8f();

  for (int kk0 = 0; kk0 < DMODEL; kk0 += 32) {
    // A tile: bf16 global rows (64 B each) -> LDS rows; async 16 B chunks
#if HAVE_ASYNC_LDS
    for (int i = tid; i < 512; i += 256) {
      const int r = i >> 2, c8 = (i & 3) * 8;
      async_ld_b128(&Actx[(size_t)(rowBase + r) * DMODEL + kk0 + c8],
                    &As[r * 32 + c8]);
    }
#else
    {
      const int t16 = tid * 16;
      const int r = t16 >> 5, c = t16 & 31;
      const __bf16* src = &Actx[(size_t)(rowBase + r) * DMODEL + kk0 + c];
      *(v4u*)&As[t16]     = *(const v4u*)src;
      *(v4u*)&As[t16 + 8] = *(const v4u*)(src + 8);
    }
#endif
    for (int i = tid; i < 2048; i += 256) {
      const int rp = i >> 7, c = i & 127;
      const int r = 2 * rp;
      BfPair pk;
      pk.h[0] = f2bf(WO[(size_t)(kk0 + r)     * DMODEL + colBase + c]);
      pk.h[1] = f2bf(WO[(size_t)(kk0 + r + 1) * DMODEL + colBase + c]);
      *(unsigned int*)&Bs[c * 32 + r] = pk.u;
    }
    tile_copy_fence();
    __syncthreads();
    v16bf a = frag_a(As, 32, wave * 16, 0, lane);
#pragma unroll
    for (int nt = 0; nt < 8; ++nt) {
      v16bf b = frag_b(Bs, 32, nt * 16, 0, lane);
      acc[nt] = wmma_bf16(a, b, acc[nt]);
    }
    __syncthreads();
  }

  const int hi = lane >> 4, nn = lane & 15;
#pragma unroll
  for (int nt = 0; nt < 8; ++nt) {
#pragma unroll
    for (int r = 0; r < 8; ++r) {
      int R = rowBase + wave * 16 + r + 8 * hi;
      int C = colBase + nt * 16 + nn;
      size_t off = (size_t)R * DMODEL + C;
      X[off] = acc[nt][r] + resid[off];
    }
  }
}

// ---------------------------------------------------------------------------
// Kernel 4: LayerNorm per row (D=1024, one block of 256 threads per row).
// ---------------------------------------------------------------------------
__global__ __launch_bounds__(256) void layernorm_kernel(
    const float* __restrict__ X, const float* __restrict__ gamma,
    const float* __restrict__ beta, float* __restrict__ out) {
  __shared__ float red[18];
  const int tid = threadIdx.x, lane = tid & 31, wave = tid >> 5;
  const size_t row = blockIdx.x;

  float4 v = ((const float4*)(X + row * DMODEL))[tid];
  float s = v.x + v.y + v.z + v.w;
  float ss = v.x * v.x + v.y * v.y + v.z * v.z + v.w * v.w;
#pragma unroll
  for (int off = 1; off < 32; off <<= 1) {
    s += __shfl_xor(s, off, 32);
    ss += __shfl_xor(ss, off, 32);
  }
  if (lane == 0) { red[wave] = s; red[8 + wave] = ss; }
  __syncthreads();
  if (tid == 0) {
    float ts = 0.0f, tss = 0.0f;
    for (int w = 0; w < 8; ++w) { ts += red[w]; tss += red[8 + w]; }
    float mean = ts * (1.0f / DMODEL);
    float var = tss * (1.0f / DMODEL) - mean * mean;
    red[16] = mean;
    red[17] = rsqrtf(var + LN_EPS);
  }
  __syncthreads();
  const float mean = red[16], rstd = red[17];
  float4 g = ((const float4*)gamma)[tid];
  float4 be = ((const float4*)beta)[tid];
  float4 o;
  o.x = (v.x - mean) * rstd * g.x + be.x;
  o.y = (v.y - mean) * rstd * g.y + be.y;
  o.z = (v.z - mean) * rstd * g.z + be.z;
  o.w = (v.w - mean) * rstd * g.w + be.w;
  ((float4*)(out + row * DMODEL))[tid] = o;
}

// ---------------------------------------------------------------------------
// Launch
// ---------------------------------------------------------------------------
extern "C" void kernel_launch(void* const* d_in, const int* in_sizes, int n_in,
                              void* d_out, int out_size, void* d_ws,
                              size_t ws_size, hipStream_t stream) {
  const float* inQ   = (const float*)d_in[0];
  const float* inK   = (const float*)d_in[1];
  const float* inV   = (const float*)d_in[2];
  // d_in[3] = attn_mask (causal, implemented analytically)
  const float* WQ    = (const float*)d_in[4];
  const float* WK    = (const float*)d_in[5];
  const float* WV    = (const float*)d_in[6];
  const float* WO    = (const float*)d_in[7];
  const float* gamma = (const float*)d_in[8];
  const float* beta  = (const float*)d_in[9];

  const size_t nElem = (size_t)BATCH * S_LEN * DMODEL;  // 8192*1024
  __bf16* Qbf = (__bf16*)d_ws;
  __bf16* Kbf = Qbf + nElem;
  __bf16* Vbf = Kbf + nElem;
  __bf16* Cbf = Vbf + nElem;
  float*  X   = (float*)(Cbf + nElem);

  float* out  = (float*)d_out;
  float* attn = out + nElem;  // attn follows out in the flat output

  qkv_proj_kernel<<<dim3(64, 8, 3), 256, 0, stream>>>(
      inQ, inK, inV, WQ, WK, WV, Qbf, Kbf, Vbf);
  attention_kernel<<<dim3(BATCH * NH, S_LEN / 128), 256, 0, stream>>>(
      Qbf, Kbf, Vbf, attn, Cbf);
  out_proj_kernel<<<dim3(64, 8), 256, 0, stream>>>(Cbf, WO, inQ, X);
  layernorm_kernel<<<dim3(BATCH * S_LEN), 256, 0, stream>>>(X, gamma, beta, out);
}
